// SynthesisLayer_43336220017234
// MI455X (gfx1250) — matrix operand order repaired and checked by reading, not compiled, not run on hardware
//
#include <hip/hip_runtime.h>

// ---------------- types ----------------
typedef _Float16 h8_t  __attribute__((ext_vector_type(8)));
typedef _Float16 h16_t __attribute__((ext_vector_type(16)));
typedef float    f8_t  __attribute__((ext_vector_type(8)));

// ---------------- problem constants ----------------
#define BATCH   4
#define CIN     512
#define COUT    512
#define HIMG    84          // input spatial
#define HPAD    88          // padded input (pad 2 each side)
#define HC      86          // conv output spatial (84 + 2*2 - 2)
#define HUP     178         // after 2x upsample filter
#define HOUT    84          // final output spatial
#define MTOT    (HC*HC)     // 7396 gemm rows per image
#define AFF_GAIN 0.04419417382415922f   // 1/sqrt(512)
#define SLOPE   0.2f
#define GAINF   1.4142135623730951f
#define CLAMPV  256.0f
#define PI_D    3.14159265358979323846

// =====================================================================
// Filter design on device (Kaiser-windowed sinc, TAPS=12, c=0.25)
// f[0..11] = fu = 2*h reversed ; f[12..23] = fd = h reversed
// =====================================================================
__device__ double bessel_i0(double x) {
    double t = x * x * 0.25;
    double term = 1.0, sum = 1.0;
    for (int k = 1; k < 64; ++k) {
        term *= t / ((double)k * (double)k);
        sum += term;
        if (term < 1e-18 * sum) break;
    }
    return sum;
}

__global__ void k_filter(float* __restrict__ f) {
    if (threadIdx.x != 0) return;
    const int T = 12;
    double nyq = 64.0, cutoff = 16.0, width = 16.0;
    double atten = 2.285 * (T - 1) * PI_D * (width / nyq) + 7.95;
    double beta;
    if (atten > 50.0)       beta = 0.1102 * (atten - 8.7);
    else if (atten > 21.0)  beta = 0.5842 * pow(atten - 21.0, 0.4) + 0.07886 * (atten - 21.0);
    else                    beta = 0.0;
    double c = cutoff / nyq;
    double i0b = bessel_i0(beta);
    double h[12];
    double sum = 0.0;
    for (int n = 0; n < T; ++n) {
        double m = (double)n - (T - 1) / 2.0;
        double xx = c * m;
        double sinc = (xx == 0.0) ? 1.0 : sin(PI_D * xx) / (PI_D * xx);
        double u = 2.0 * n / (T - 1) - 1.0;
        double arg = 1.0 - u * u; if (arg < 0.0) arg = 0.0;
        double win = bessel_i0(beta * sqrt(arg)) / i0b;
        h[n] = c * sinc * win;
        sum += h[n];
    }
    for (int n = 0; n < T; ++n) h[n] /= sum;
    for (int t = 0; t < T; ++t) {
        f[t]      = (float)(2.0 * h[T - 1 - t]);  // fu (upsample, gain UP)
        f[12 + t] = (float)(h[T - 1 - t]);        // fd (downsample)
    }
}

// =====================================================================
// s[b,c] = AFF_GAIN * <w_latent[b,:], aff_w[c,:]> + aff_b[c]
// =====================================================================
__global__ void k_affine(const float* __restrict__ wl, const float* __restrict__ aw,
                         const float* __restrict__ ab, float* __restrict__ s) {
    int idx = blockIdx.x * blockDim.x + threadIdx.x;
    if (idx >= BATCH * CIN) return;
    int b = idx >> 9, c = idx & 511;
    float acc = 0.f;
    for (int k = 0; k < 512; ++k) acc += wl[b * 512 + k] * aw[c * 512 + k];
    s[idx] = acc * AFF_GAIN + ab[c];
}

// sn = s * rsqrt(mean(s^2))  (mean over all 2048 values)
__global__ void k_norm_s(const float* __restrict__ s, float* __restrict__ sn) {
    __shared__ float red[256];
    float acc = 0.f;
    for (int i = threadIdx.x; i < BATCH * CIN; i += 256) { float v = s[i]; acc += v * v; }
    red[threadIdx.x] = acc; __syncthreads();
    for (int off = 128; off > 0; off >>= 1) {
        if ((int)threadIdx.x < off) red[threadIdx.x] += red[threadIdx.x + off];
        __syncthreads();
    }
    float scale = rsqrtf(red[0] / (float)(BATCH * CIN));
    for (int i = threadIdx.x; i < BATCH * CIN; i += 256) sn[i] = s[i] * scale;
}

// =====================================================================
// Per out-channel: wn = w * rsqrt(mean(w^2)), w2[o,ci] = sum_r wn^2
// wn stored f16 in WMMA-B friendly layout: [r][ci/32][o][ci%32]
// =====================================================================
__global__ void k_wprep(const float* __restrict__ cw, _Float16* __restrict__ wn,
                        float* __restrict__ w2) {
    int o = blockIdx.x;
    __shared__ float red[128];
    const float* w = cw + (size_t)o * (CIN * 9);
    float acc = 0.f;
    for (int i = threadIdx.x; i < CIN * 9; i += 128) { float v = w[i]; acc += v * v; }
    red[threadIdx.x] = acc; __syncthreads();
    for (int off = 64; off > 0; off >>= 1) {
        if ((int)threadIdx.x < off) red[threadIdx.x] += red[threadIdx.x + off];
        __syncthreads();
    }
    float rn = rsqrtf(red[0] / (float)(CIN * 9));
    for (int ci = threadIdx.x; ci < CIN; ci += 128) {
        float s2 = 0.f;
        for (int r = 0; r < 9; ++r) {
            float v = w[ci * 9 + r] * rn;
            s2 += v * v;
            wn[(((size_t)r * 16 + (ci >> 5)) * COUT + o) * 32 + (ci & 31)] = (_Float16)v;
        }
        w2[o * CIN + ci] = s2;
    }
}

// dcoef[b,o] = rsqrt(sum_ci sn[b,ci]^2 * w2[o,ci] + 1e-8)
__global__ void k_dcoef(const float* __restrict__ sn, const float* __restrict__ w2,
                        float* __restrict__ dc) {
    int idx = blockIdx.x * blockDim.x + threadIdx.x;
    if (idx >= BATCH * COUT) return;
    int b = idx >> 9, o = idx & 511;
    float acc = 0.f;
    for (int ci = 0; ci < CIN; ++ci) {
        float sv = sn[b * CIN + ci];
        acc += sv * sv * w2[o * CIN + ci];
    }
    dc[idx] = rsqrtf(acc + 1e-8f);
}

// =====================================================================
// Pack xm = x * sn -> f16, channels-last, zero-padded [b][88][88][512]
// LDS transpose for coalesced read (w-fastest) and write (c-fastest)
// =====================================================================
__global__ void k_pack_x(const float* __restrict__ x, const float* __restrict__ sn,
                         _Float16* __restrict__ xm) {
    int hp = blockIdx.x;          // 0..87
    int c0 = blockIdx.y * 64;
    int b  = blockIdx.z;
    __shared__ _Float16 lds[HPAD * 65];
    int h = hp - 2;
    for (int idx = threadIdx.x; idx < 64 * HPAD; idx += 256) {
        int cl = idx / HPAD, wp = idx % HPAD;
        int w = wp - 2;
        float v = 0.f;
        if ((unsigned)h < (unsigned)HIMG && (unsigned)w < (unsigned)HIMG)
            v = x[(((size_t)b * CIN + c0 + cl) * HIMG + h) * HIMG + w] * sn[b * CIN + c0 + cl];
        lds[wp * 65 + cl] = (_Float16)v;
    }
    __syncthreads();
    for (int idx = threadIdx.x; idx < HPAD * 64; idx += 256) {
        int wp = idx >> 6, cl = idx & 63;
        xm[(((size_t)b * HPAD + hp) * HPAD + wp) * CIN + c0 + cl] = lds[wp * 65 + cl];
    }
}

// =====================================================================
// Implicit-GEMM 3x3 conv via v_wmma_f32_16x16x32_f16.
// One wave per block; wave tile = 32 M (spatial) x 64 N (cout).
// K loop: 16 chunks of 32 input channels x 9 taps -> 1152 WMMAs/wave.
// A frag (16x32 f16): lane row m=lane%16; halves = K {hi*8..+7, 16+hi*8..+7}
// B frag (32x16 f16): lane col n=lane%16; halves = K {hi*8..+7, 16+hi*8..+7}
// Epilogue: y0 = acc*dcoef + bias, channels-last f32 [b][86][86][512]
// =====================================================================
__global__ void __launch_bounds__(32)
k_conv(const _Float16* __restrict__ xm, const _Float16* __restrict__ wn,
       const float* __restrict__ dcoef, const float* __restrict__ cb,
       float* __restrict__ y0) {
    const int lane = threadIdx.x;
    const int nl = lane & 15;
    const int hi = lane >> 4;
    const int m0 = blockIdx.x * 32;
    const int n0 = blockIdx.y * 64;
    const int b  = blockIdx.z;

    int hh[2], ww[2];
    #pragma unroll
    for (int s = 0; s < 2; ++s) {
        int m = m0 + s * 16 + nl;
        if (m > MTOT - 1) m = MTOT - 1;      // clamp loads; stores masked later
        hh[s] = m / HC; ww[s] = m % HC;
    }

    f8_t acc[2][4] = {};

    for (int cic = 0; cic < 16; ++cic) {
        const int ci0 = cic * 32;
        for (int r = 0; r < 9; ++r) {
            const int ky = r / 3, kx = r % 3;
            h16_t a[2];
            #pragma unroll
            for (int s = 0; s < 2; ++s) {
                const _Float16* p =
                    xm + (((size_t)b * HPAD + hh[s] + ky) * HPAD + (ww[s] + kx)) * CIN + ci0;
                h8_t lo = *(const h8_t*)(p + hi * 8);
                h8_t up = *(const h8_t*)(p + 16 + hi * 8);
                a[s] = __builtin_shufflevector(lo, up, 0,1,2,3,4,5,6,7,8,9,10,11,12,13,14,15);
            }
            #pragma unroll
            for (int t = 0; t < 4; ++t) {
                const int o = n0 + t * 16 + nl;
                const _Float16* q =
                    wn + (((size_t)r * 16 + cic) * COUT + o) * 32;
                h8_t lo = *(const h8_t*)(q + hi * 8);
                h8_t up = *(const h8_t*)(q + 16 + hi * 8);
                h16_t bf = __builtin_shufflevector(lo, up, 0,1,2,3,4,5,6,7,8,9,10,11,12,13,14,15);
                #pragma unroll
                for (int s = 0; s < 2; ++s)
                    acc[s][t] = __builtin_amdgcn_wmma_f32_16x16x32_f16(
                        false, a[s], false, bf, (short)0, acc[s][t], false, false);
            }
        }
    }

    // epilogue: demodulate + bias, store channels-last
    for (int t = 0; t < 4; ++t) {
        const int o = n0 + t * 16 + nl;
        const float dc  = dcoef[b * COUT + o];
        const float bia = cb[o];
        for (int s = 0; s < 2; ++s) {
            #pragma unroll
            for (int rr = 0; rr < 8; ++rr) {
                int m = m0 + s * 16 + hi * 8 + rr;
                if (m < MTOT) {
                    int h = m / HC, w = m % HC;
                    y0[(((size_t)b * HC + h) * HC + w) * COUT + o] = acc[s][t][rr] * dc + bia;
                }
            }
        }
    }
}

// =====================================================================
// Upsample along H: y1[b][h1][w][c], h1 in [0,178)
// =====================================================================
__global__ void k_up_h(const float* __restrict__ y0, const float* __restrict__ fw,
                       float* __restrict__ y1) {
    int h1 = blockIdx.x;
    int b  = blockIdx.y;
    int p  = (h1 + 1) & 1;
    for (int idx = threadIdx.x; idx < HC * CIN; idx += 256) {
        int w = idx >> 9, c = idx & 511;
        float acc = 0.f;
        for (int t = p; t < 12; t += 2) {
            int j = h1 + t - 9;
            if (j >= 0) {
                int src = j >> 1;
                if (src < HC)
                    acc += fw[t] * y0[(((size_t)b * HC + src) * HC + w) * COUT + c];
            }
        }
        y1[(((size_t)b * HUP + h1) * HC + w) * COUT + c] = acc;
    }
}

// =====================================================================
// Fused along W: upsample -> leaky*gain -> clamp -> downsample
// block = (h1, 32-channel chunk, b); row staged in LDS
// =====================================================================
__global__ void k_w_pipe(const float* __restrict__ y1, const float* __restrict__ fw,
                         float* __restrict__ y2) {
    int h1 = blockIdx.x;
    int c0 = blockIdx.y * 32;
    int b  = blockIdx.z;
    const float* fu = fw;
    const float* fd = fw + 12;
    __shared__ float in[HC * 32];
    __shared__ float full[HUP * 32];
    for (int idx = threadIdx.x; idx < HC * 32; idx += 256) {
        int w = idx >> 5, c = idx & 31;
        in[idx] = y1[(((size_t)b * HUP + h1) * HC + w) * COUT + c0 + c];
    }
    __syncthreads();
    for (int idx = threadIdx.x; idx < HUP * 32; idx += 256) {
        int ww = idx >> 5, c = idx & 31;
        int p = (ww + 1) & 1;
        float acc = 0.f;
        for (int t = p; t < 12; t += 2) {
            int j = ww + t - 9;
            if (j >= 0) { int s = j >> 1; if (s < HC) acc += fu[t] * in[s * 32 + c]; }
        }
        acc = (acc < 0.f ? acc * SLOPE : acc) * GAINF;
        acc = fminf(fmaxf(acc, -CLAMPV), CLAMPV);
        full[idx] = acc;
    }
    __syncthreads();
    for (int idx = threadIdx.x; idx < HOUT * 32; idx += 256) {
        int wo = idx >> 5, c = idx & 31;
        float acc = 0.f;
        #pragma unroll
        for (int t = 0; t < 12; ++t) acc += fd[t] * full[(2 * wo + t) * 32 + c];
        y2[(((size_t)b * HUP + h1) * HOUT + wo) * COUT + c0 + c] = acc;
    }
}

// =====================================================================
// Downsample along H + transpose to NCHW output via LDS
// =====================================================================
__global__ void k_down_h(const float* __restrict__ y2, const float* __restrict__ fw,
                         float* __restrict__ out) {
    int ho = blockIdx.x;
    int c0 = blockIdx.y * 64;
    int b  = blockIdx.z;
    const float* fd = fw + 12;
    __shared__ float tile[64 * 85];
    for (int idx = threadIdx.x; idx < HOUT * 64; idx += 256) {
        int cl = idx & 63, wo = idx >> 6;
        float acc = 0.f;
        #pragma unroll
        for (int t = 0; t < 12; ++t)
            acc += fd[t] * y2[(((size_t)b * HUP + 2 * ho + t) * HOUT + wo) * COUT + c0 + cl];
        tile[cl * 85 + wo] = acc;
    }
    __syncthreads();
    for (int idx = threadIdx.x; idx < 64 * HOUT; idx += 256) {
        int wo = idx % HOUT, cl = idx / HOUT;
        out[(((size_t)b * COUT + c0 + cl) * HOUT + ho) * HOUT + wo] = tile[cl * 85 + wo];
    }
}

// =====================================================================
extern "C" void kernel_launch(void* const* d_in, const int* in_sizes, int n_in,
                              void* d_out, int out_size, void* d_ws, size_t ws_size,
                              hipStream_t stream) {
    (void)in_sizes; (void)n_in; (void)out_size; (void)ws_size;
    const float* x  = (const float*)d_in[0];
    const float* wl = (const float*)d_in[1];
    const float* aw = (const float*)d_in[2];
    const float* ab = (const float*)d_in[3];
    const float* cw = (const float*)d_in[4];
    const float* cb = (const float*)d_in[5];
    float* out = (float*)d_out;

    char* ws = (char*)d_ws;
    size_t off = 0;
    auto alloc = [&](size_t bytes) { size_t o = off; off = (off + bytes + 255) & ~(size_t)255; return o; };

    float*    f_filt = (float*)(ws + alloc(24 * 4));
    float*    s_buf  = (float*)(ws + alloc((size_t)BATCH * CIN * 4));
    float*    sn_buf = (float*)(ws + alloc((size_t)BATCH * CIN * 4));
    float*    dc_buf = (float*)(ws + alloc((size_t)BATCH * COUT * 4));
    float*    w2_buf = (float*)(ws + alloc((size_t)COUT * CIN * 4));
    _Float16* wn_buf = (_Float16*)(ws + alloc((size_t)9 * CIN * COUT * 2));

    // lifetime aliasing: xm (31.7MB) lives inside y1's slot (125.4MB);
    // y0 (60.6MB) lives inside y2's slot (122.5MB).
    char* y1_base = ws + alloc((size_t)BATCH * HUP * HC * COUT * 4);
    char* y2_base = ws + alloc((size_t)BATCH * HUP * HOUT * COUT * 4);
    _Float16* xm_buf = (_Float16*)y1_base;
    float*    y1_buf = (float*)y1_base;
    float*    y0_buf = (float*)y2_base;
    float*    y2_buf = (float*)y2_base;

    k_filter<<<1, 32, 0, stream>>>(f_filt);
    k_affine<<<8, 256, 0, stream>>>(wl, aw, ab, s_buf);
    k_norm_s<<<1, 256, 0, stream>>>(s_buf, sn_buf);
    k_wprep<<<COUT, 128, 0, stream>>>(cw, wn_buf, w2_buf);
    k_dcoef<<<8, 256, 0, stream>>>(sn_buf, w2_buf, dc_buf);
    k_pack_x<<<dim3(HPAD, CIN / 64, BATCH), 256, 0, stream>>>(x, sn_buf, xm_buf);
    k_conv<<<dim3((MTOT + 31) / 32, COUT / 64, BATCH), 32, 0, stream>>>(
        xm_buf, wn_buf, dc_buf, cb, y0_buf);
    k_up_h<<<dim3(HUP, BATCH), 256, 0, stream>>>(y0_buf, f_filt, y1_buf);
    k_w_pipe<<<dim3(HUP, CIN / 32, BATCH), 256, 0, stream>>>(y1_buf, f_filt, y2_buf);
    k_down_h<<<dim3(HOUT, COUT / 64, BATCH), 256, 0, stream>>>(y2_buf, f_filt, out);
}
